// Seq2Seq_78520592106054
// MI455X (gfx1250) — matrix operand order repaired and checked by reading, not compiled, not run on hardware
//
#include <hip/hip_runtime.h>

typedef _Float16 v16h __attribute__((ext_vector_type(16)));
typedef float    v8f  __attribute__((ext_vector_type(8)));

constexpr int N  = 16384;
constexpr int H  = 128;
constexpr long NH = (long)N * H;

// Workspace layout (floats):
//   [0, N)        xnorm[i] = ||bi_enc_i||^2
//   [N, 2N)       diag[i]  = ||bi_enc_i - bi_dec_i||^2
//   [2N, 2N+128)  S_y[h]   = sum_i bi_dec[i][h]
//   2N+128        sum_ynorm
//   2N+129        sum_diag
//   2N+130        nce accumulator
#define WS_XNORM 0
#define WS_DIAG  (N)
#define WS_SY    (2*N)
#define WS_SUMYN (2*N + 128)
#define WS_SUMDG (2*N + 129)
#define WS_NCE   (2*N + 130)

__global__ void zero_acc_kernel(float* __restrict__ ws) {
    int t = threadIdx.x;
    if (t < 131) ws[2*N + t] = 0.0f;   // S_y[128] + 3 scalars
}

// Streams enc/dec once: writes bi_enc (x2), bi_dec, and reduces
// per-row xnorm/diag + global S_y / sum_ynorm / sum_diag.
// 256 blocks x 256 threads; each block owns 64 rows, 2 rows per iteration.
__global__ __launch_bounds__(256) void fuse_reduce_kernel(
        const float* __restrict__ enc, const float* __restrict__ dec,
        float* __restrict__ out, float* __restrict__ ws) {
    __shared__ float part[8][3];     // per-wave partials (vx, vy, vd)
    __shared__ float sycomb[128];
    __shared__ float blkAcc[2][2];   // [row-slot][{ynorm, diag}]

    const int tid  = threadIdx.x;
    const int c    = tid & 127;      // column 0..127
    const int rl   = tid >> 7;       // row slot 0/1
    const int wid  = tid >> 5;       // wave id 0..7 (wave32)
    const int lane = tid & 31;

    if (tid < 4) (&blkAcc[0][0])[tid] = 0.0f;
    __syncthreads();

    float syacc = 0.0f;
    const int rowBase = blockIdx.x * 64;
    float* bi_enc0 = out;
    float* bi_enc1 = out + NH;
    float* bi_dec  = out + 2 * NH;

    for (int it = 0; it < 32; ++it) {
        const int row = rowBase + it * 2 + rl;
        const float* e = enc + (size_t)row * 256;   // [N, D=2, H=128]
        const float* d = dec + (size_t)row * 256;
        // CDNA5 speculative prefetch (global_prefetch_b8); OOB is silently dropped.
        __builtin_prefetch(e + 512 + c, 0, 1);
        __builtin_prefetch(d + 512 + c, 0, 1);

        const float x = e[c] + e[c + 128];          // sum over direction axis
        const float y = d[c] + d[c + 128];
        bi_enc0[(size_t)row * 128 + c] = x;
        bi_enc1[(size_t)row * 128 + c] = x;
        bi_dec [(size_t)row * 128 + c] = y;

        syacc += y;
        const float dd = x - y;
        float vx = x * x, vy = y * y, vd = dd * dd;
        #pragma unroll
        for (int o = 16; o > 0; o >>= 1) {          // wave32 reduction
            vx += __shfl_down(vx, o, 32);
            vy += __shfl_down(vy, o, 32);
            vd += __shfl_down(vd, o, 32);
        }
        if (lane == 0) { part[wid][0] = vx; part[wid][1] = vy; part[wid][2] = vd; }
        __syncthreads();
        if (c == 0) {
            float sx = 0.0f, sy2 = 0.0f, sd = 0.0f;
            const int w0 = rl * 4;
            #pragma unroll
            for (int w = w0; w < w0 + 4; ++w) {
                sx += part[w][0]; sy2 += part[w][1]; sd += part[w][2];
            }
            ws[WS_XNORM + row] = sx;
            ws[WS_DIAG  + row] = sd;
            blkAcc[rl][0] += sy2;
            blkAcc[rl][1] += sd;
        }
        __syncthreads();
    }

    // Fold the two row-slot column accumulators, then one atomic per column per block.
    if (rl == 0) sycomb[c] = syacc;
    __syncthreads();
    if (rl == 1) sycomb[c] += syacc;
    __syncthreads();
    if (rl == 0) atomicAdd(&ws[WS_SY + c], sycomb[c]);
    if (tid == 0) {
        atomicAdd(&ws[WS_SUMYN], blkAcc[0][0] + blkAcc[1][0]);
        atomicAdd(&ws[WS_SUMDG], blkAcc[0][1] + blkAcc[1][1]);
    }
}

// One wave (32 lanes, EXEC all-1s) per 16-row tile. Computes dot[i] = x_i . S_y
// with v_wmma_f32_16x16x32_f16 (B = S_y replicated across 16 columns), then the
// clamped row loss. 1024 blocks x 32 threads.
__global__ __launch_bounds__(32) void wmma_rowloss_kernel(
        const float* __restrict__ X, float* __restrict__ ws) {
    const int lane = threadIdx.x;
    const int half = lane >> 4;          // 0: lanes 0-15, 1: lanes 16-31
    const int mrow = lane & 15;
    const int R    = blockIdx.x * 16;

    const float* xrow = X + (size_t)(R + mrow) * 128;
    const float* Sy   = ws + WS_SY;

    v8f acc = {};
    #pragma unroll
    for (int kb = 0; kb < 128; kb += 32) {
        v16h a, b;
        // 16-bit A 16x32 layout: lanes 0-15 hold K {0..7, 16..23}, lanes 16-31 hold {8..15, 24..31}
        #pragma unroll
        for (int e = 0; e < 8; ++e) {
            a[e]     = (_Float16)xrow[kb + half * 8 + e];
            a[8 + e] = (_Float16)xrow[kb + 16 + half * 8 + e];
        }
        // B 32x16: column = lane&15; lanes 0-15 carry K {0..7, 16..23}, lanes 16-31 carry K {8..15, 24..31}.
        // B[k][n] = Sy[kb+k] for all n (replicated), so every D column equals the dot.
        #pragma unroll
        for (int e = 0; e < 8; ++e) {
            b[e]     = (_Float16)Sy[kb + half * 8 + e];
            b[8 + e] = (_Float16)Sy[kb + 16 + half * 8 + e];
        }
        acc = __builtin_amdgcn_wmma_f32_16x16x32_f16(
                false, a, false, b, (short)0, acc, false, false);
    }

    // D layout: lane 0 -> rows 0..7 (VGPR v = row v), lane 16 -> rows 8..15.
    if (mrow == 0) {
        const float sum_ynorm = ws[WS_SUMYN];
        float local = 0.0f;
        #pragma unroll
        for (int v = 0; v < 8; ++v) {
            const int r = R + half * 8 + v;
            const float dot    = acc[v];
            const float rowsum = (float)N * ws[WS_XNORM + r] + sum_ynorm - 2.0f * dot;
            const float rloss  = 5.0f - rowsum + 10.0f * ws[WS_DIAG + r];
            local += fmaxf(rloss, 0.0f);
        }
        atomicAdd(&ws[WS_NCE], local);
    }
}

__global__ void finalize_kernel(const float* __restrict__ ws, float* __restrict__ out) {
    if (threadIdx.x == 0) {
        out[3 * NH]     = ws[WS_NCE];
        out[3 * NH + 1] = -ws[WS_SUMDG] / (float)N;
    }
}

extern "C" void kernel_launch(void* const* d_in, const int* in_sizes, int n_in,
                              void* d_out, int out_size, void* d_ws, size_t ws_size,
                              hipStream_t stream) {
    (void)in_sizes; (void)n_in; (void)out_size; (void)ws_size;
    const float* enc = (const float*)d_in[0];
    const float* dec = (const float*)d_in[1];
    float* out = (float*)d_out;
    float* ws  = (float*)d_ws;

    zero_acc_kernel   <<<1,     256, 0, stream>>>(ws);
    fuse_reduce_kernel<<<256,   256, 0, stream>>>(enc, dec, out, ws);
    wmma_rowloss_kernel<<<N / 16, 32, 0, stream>>>(out, ws);
    finalize_kernel   <<<1,       1, 0, stream>>>(ws, out);
}
